// bRNNC_15642270892456
// MI455X (gfx1250) — matrix operand order repaired and checked by reading, not compiled, not run on hardware
//
#include <hip/hip_runtime.h>
#include <hip/hip_bf16.h>

// ---------------------------------------------------------------------------
// Bidirectional LSTM (B=64, S=1024, D=128, H=256) + output linear (O=64) + mask
// CDNA5 / gfx1250: wave32, v_wmma_f32_16x16x32_bf16, async LDS load + store
// ---------------------------------------------------------------------------

#define B_DIM   64
#define S_LEN   1024
#define D_IN    128
#define H_DIM   256
#define G4H     1024      // 4*H
#define KA      384       // H + D (fused recurrent+input projection K)
#define KA_PAD  392       // padded LDS row stride in elements (784 bytes)
#define O_DIM   64
#define K_OUT   512       // 2*H

typedef __attribute__((ext_vector_type(16))) __bf16 v16bf;
typedef __attribute__((ext_vector_type(8)))  float  v8f;

__device__ __forceinline__ float sigmoidf_(float x) {
    return 1.0f / (1.0f + __expf(-x));
}
__device__ __forceinline__ float tanhf_(float x) {
    float e = __expf(-2.0f * x);
    return (1.0f - e) / (1.0f + e);
}

// ----------------------------- prep kernels --------------------------------

__global__ void cvt_f32_bf16(const float* __restrict__ in,
                             __bf16* __restrict__ out, int n) {
    int i = blockIdx.x * blockDim.x + threadIdx.x;
    if (i < n) out[i] = (__bf16)in[i];
}

// Wcat[n][k] : k<256 -> W_hh[n][k], k>=256 -> W_ih[n][k-256]   (bf16, row-major)
__global__ void build_wcat(const float* __restrict__ Whh,
                           const float* __restrict__ Wih,
                           __bf16* __restrict__ Wcat) {
    int i = blockIdx.x * blockDim.x + threadIdx.x;
    if (i >= G4H * KA) return;
    int n = i / KA, k = i % KA;
    float v = (k < H_DIM) ? Whh[n * H_DIM + k] : Wih[n * D_IN + (k - H_DIM)];
    Wcat[(size_t)n * KA + k] = (__bf16)v;
}

// ----------------------------- recurrent scan ------------------------------
// grid = 2 (dir), block = 1024 threads (32 waves).
// A-matrix in LDS: [64 rows (batch)] x [KA_PAD cols] bf16, cols 0..255 = h,
// cols 256..383 = x_s. Wave w: m-tile = w&3 (16 batch rows), j-group = w>>2
// (32 hidden columns), all 4 gates for those columns -> 8 WMMA tiles,
// processed as two passes of 4 tiles to bound VGPR pressure.

// Async stage of x_s into LDS A cols 256..383 (64 rows x 128 cols bf16,
// 1024 threads x 16 bytes). Tracked with ASYNCcnt.
__device__ __forceinline__ void stage_x_async(const __bf16* __restrict__ xb,
                                              __bf16* __restrict__ Aarr,
                                              int tid, int s) {
    int b = tid >> 4, seg = tid & 15;
    unsigned long long ga =
        (unsigned long long)(xb + ((size_t)b * S_LEN + s) * D_IN + seg * 8);
    unsigned lda = (unsigned)(uintptr_t)&Aarr[b * KA_PAD + H_DIM + seg * 8];
    asm volatile("global_load_async_to_lds_b128 %0, %1, off"
                 :: "v"(lda), "v"(ga) : "memory");
}

// Async store of h (LDS A cols 0..255) to global hs[b][s][:].
// 1024 threads x 32 bytes (two b128, offset: applies to both LDS & global).
__device__ __forceinline__ void store_h_async(__bf16* __restrict__ hs,
                                              const __bf16* __restrict__ Aarr,
                                              int tid, int s) {
    int b = tid >> 4, seg = tid & 15;
    unsigned l0 = (unsigned)(uintptr_t)&Aarr[b * KA_PAD + seg * 16];
    unsigned long long g0 =
        (unsigned long long)(hs + ((size_t)b * S_LEN + s) * H_DIM + seg * 16);
    asm volatile("global_store_async_from_lds_b128 %0, %1, off\n\t"
                 "global_store_async_from_lds_b128 %0, %1, off offset:16"
                 :: "v"(g0), "v"(l0) : "memory");
}

__global__ __launch_bounds__(1024)
void lstm_scan_kernel(const __bf16* __restrict__ xb,
                      const __bf16* __restrict__ WcatF,
                      const __bf16* __restrict__ WcatB,
                      const float*  __restrict__ bF,
                      const float*  __restrict__ bB,
                      const float*  __restrict__ h0F,
                      const float*  __restrict__ c0F,
                      const float*  __restrict__ h0B,
                      const float*  __restrict__ c0B,
                      __bf16* __restrict__ hsF,
                      __bf16* __restrict__ hsB) {
    const int dir = blockIdx.x;
    const __bf16* __restrict__ Wcat = dir ? WcatB : WcatF;
    const float*  __restrict__ bias = dir ? bB : bF;
    const float*  __restrict__ h0   = dir ? h0B : h0F;
    const float*  __restrict__ c0   = dir ? c0B : c0F;
    __bf16* __restrict__ hs         = dir ? hsB : hsF;

    __shared__ __align__(16) __bf16 A[B_DIM * KA_PAD];

    const int tid  = threadIdx.x;
    const int wave = tid >> 5;
    const int lane = tid & 31;
    const int half = lane >> 4;
    const int l15  = lane & 15;
    const int mbase = (wave & 3) * 16;     // batch-row tile
    const int jbase = (wave >> 2) * 32;    // hidden-column group (2 j-tiles)

    // ---- prologue: stage x_{s0} (async) and h0 (cols 0..255) ----
    stage_x_async(xb, A, tid, dir ? (S_LEN - 1) : 0);
    {
        int b = tid >> 4, seg = tid & 15;  // 16 floats each
        const float* src = h0 + b * H_DIM + seg * 16;
        __bf16* dst = &A[b * KA_PAD + seg * 16];
        #pragma unroll
        for (int i = 0; i < 16; ++i) dst[i] = (__bf16)src[i];
    }

    // ---- per-wave persistent state: c in registers, bias in registers ----
    float cacc[2][8];
    float bgate[2][4];
    #pragma unroll
    for (int jt = 0; jt < 2; ++jt) {
        int j = jbase + jt * 16 + l15;
        #pragma unroll
        for (int r = 0; r < 8; ++r) {
            int m = mbase + r + half * 8;
            cacc[jt][r] = c0[m * H_DIM + j];
        }
        #pragma unroll
        for (int g = 0; g < 4; ++g)
            bgate[jt][g] = bias[g * H_DIM + j];
    }

    // ---- sequential scan ----
    for (int t = 0; t < S_LEN; ++t) {
        const int s = dir ? (S_LEN - 1 - t) : t;

        asm volatile("s_wait_asynccnt 0x0" ::: "memory"); // x_s / h drained
        __syncthreads();                                   // A (h + x_s) ready

        // Opaque integer offset: blocks LICM of the weight stream while
        // preserving global-pointer provenance (=> global_load_b128, not flat).
        size_t wofs = 0;
        asm volatile("" : "+s"(wofs));
        const __bf16* __restrict__ Wt = Wcat + wofs;

        __bf16 hreg[2][8];
        #pragma unroll
        for (int jt = 0; jt < 2; ++jt) {
            const int jcol = jbase + jt * 16 + l15;

            v8f acc[4];
            #pragma unroll
            for (int g = 0; g < 4; ++g)
                acc[g] = (v8f){0.f,0.f,0.f,0.f,0.f,0.f,0.f,0.f};

            for (int kc = 0; kc < KA / 32; ++kc) {
                const int kbase = kc * 32;
                v16bf afrag;
                {
                    uint4* ap = (uint4*)&afrag;
                    const __bf16* arow =
                        &A[(mbase + l15) * KA_PAD + kbase + half * 8];
                    ap[0] = *(const uint4*)(arow);
                    ap[1] = *(const uint4*)(arow + 16);   // K + 16
                }
                v16bf bfr[4];
                #pragma unroll
                for (int g = 0; g < 4; ++g) {
                    uint4* bp = (uint4*)&bfr[g];
                    const uint4* wp = (const uint4*)(
                        Wt + (size_t)(g * H_DIM + jcol) * KA + kbase + half * 8);
                    bp[0] = wp[0];
                    bp[1] = wp[2];
                }
                #pragma unroll
                for (int g = 0; g < 4; ++g)
                    acc[g] = __builtin_amdgcn_wmma_f32_16x16x32_bf16(
                        false, afrag, false, bfr[g], (short)0, acc[g],
                        false, false);
            }

            // ---- pointwise LSTM update, fully in registers ----
            #pragma unroll
            for (int r = 0; r < 8; ++r) {
                float iv = sigmoidf_(acc[0][r] + bgate[jt][0]);
                float fv = sigmoidf_(acc[1][r] + bgate[jt][1]);
                float gv = tanhf_   (acc[2][r] + bgate[jt][2]);
                float ov = sigmoidf_(acc[3][r] + bgate[jt][3]);
                float c  = fv * cacc[jt][r] + iv * gv;
                cacc[jt][r] = c;
                hreg[jt][r] = (__bf16)(ov * tanhf_(c));
            }
        }

        __syncthreads();   // all waves done reading A this step

        // ---- write h_new into LDS (next step's A operand) ----
        #pragma unroll
        for (int jt = 0; jt < 2; ++jt) {
            int j = jbase + jt * 16 + l15;
            #pragma unroll
            for (int r = 0; r < 8; ++r) {
                int m = mbase + r + half * 8;   // batch index
                A[m * KA_PAD + j] = hreg[jt][r];
            }
        }

        __syncthreads();   // h region complete in LDS

        // ---- async: LDS h -> global hs, and stage next x_s into LDS ----
        store_h_async(hs, A, tid, s);
        if (t + 1 < S_LEN)
            stage_x_async(xb, A, tid, dir ? (S_LEN - 2 - t) : (t + 1));
    }
}

// ----------------------------- output GEMM ---------------------------------
// out[m][o] = (sum_k hsF[m][k]*Wout[o][k] + hsB[m][k]*Wout[o][256+k] + b[o])
//             * mask[m],   m = b*S+s,   M=65536, N=64, K=512
// block = 256 threads (8 waves), wave -> one 16-row m-tile, all 4 n-tiles.

__global__ __launch_bounds__(256)
void out_gemm_kernel(const __bf16* __restrict__ hsF,
                     const __bf16* __restrict__ hsB,
                     const __bf16* __restrict__ WoutB,  // [64][512] bf16
                     const float*  __restrict__ b_out,
                     const float*  __restrict__ mask,
                     float* __restrict__ out) {
    const int wave = threadIdx.x >> 5;
    const int lane = threadIdx.x & 31;
    const int half = lane >> 4;
    const int l15  = lane & 15;
    const int mbase = blockIdx.x * 128 + wave * 16;

    v8f acc[4];
    #pragma unroll
    for (int nt = 0; nt < 4; ++nt)
        acc[nt] = (v8f){0.f,0.f,0.f,0.f,0.f,0.f,0.f,0.f};

    for (int kc = 0; kc < K_OUT / 32; ++kc) {
        const int kbase = kc * 32;
        const __bf16* __restrict__ src = (kbase < H_DIM) ? hsF : hsB;
        const int kk = kbase & (H_DIM - 1);

        v16bf afrag;
        {
            uint4* ap = (uint4*)&afrag;
            const __bf16* arow =
                src + (size_t)(mbase + l15) * H_DIM + kk + half * 8;
            ap[0] = *(const uint4*)(arow);
            ap[1] = *(const uint4*)(arow + 16);
        }
        v16bf bfr[4];
        #pragma unroll
        for (int nt = 0; nt < 4; ++nt) {
            uint4* bp = (uint4*)&bfr[nt];
            const uint4* wp = (const uint4*)(
                WoutB + (size_t)(nt * 16 + l15) * K_OUT + kbase + half * 8);
            bp[0] = wp[0];
            bp[1] = wp[2];
        }
        #pragma unroll
        for (int nt = 0; nt < 4; ++nt)
            acc[nt] = __builtin_amdgcn_wmma_f32_16x16x32_bf16(
                false, afrag, false, bfr[nt], (short)0, acc[nt], false, false);
    }

    #pragma unroll
    for (int nt = 0; nt < 4; ++nt) {
        const int o = nt * 16 + l15;
        const float bo = b_out[o];
        #pragma unroll
        for (int r = 0; r < 8; ++r) {
            const int m = mbase + r + half * 8;
            out[(size_t)m * O_DIM + o] = (acc[nt][r] + bo) * mask[m];
        }
    }
}

// ----------------------------- host launcher -------------------------------

extern "C" void kernel_launch(void* const* d_in, const int* in_sizes, int n_in,
                              void* d_out, int out_size, void* d_ws,
                              size_t ws_size, hipStream_t stream) {
    const float* x      = (const float*)d_in[0];
    const float* mask   = (const float*)d_in[1];
    const float* h0_f   = (const float*)d_in[2];
    const float* c0_f   = (const float*)d_in[3];
    const float* h0_b   = (const float*)d_in[4];
    const float* c0_b   = (const float*)d_in[5];
    const float* W_ih_f = (const float*)d_in[6];
    const float* W_hh_f = (const float*)d_in[7];
    const float* b_f    = (const float*)d_in[8];
    const float* W_ih_b = (const float*)d_in[9];
    const float* W_hh_b = (const float*)d_in[10];
    const float* b_b    = (const float*)d_in[11];
    const float* W_out  = (const float*)d_in[12];
    const float* b_out  = (const float*)d_in[13];
    float* out = (float*)d_out;

    // workspace layout (bytes)
    char* ws = (char*)d_ws;
    size_t off = 0;
    __bf16* xb    = (__bf16*)(ws + off); off += (size_t)B_DIM * S_LEN * D_IN * 2;   // 16 MB
    __bf16* wcatF = (__bf16*)(ws + off); off += (size_t)G4H * KA * 2;               // 768 KB
    __bf16* wcatB = (__bf16*)(ws + off); off += (size_t)G4H * KA * 2;               // 768 KB
    __bf16* woutB = (__bf16*)(ws + off); off += (size_t)O_DIM * K_OUT * 2;          // 64 KB
    __bf16* hsF   = (__bf16*)(ws + off); off += (size_t)B_DIM * S_LEN * H_DIM * 2;  // 32 MB
    __bf16* hsB   = (__bf16*)(ws + off); off += (size_t)B_DIM * S_LEN * H_DIM * 2;  // 32 MB

    // 1) precision conversion / weight packing
    {
        int n = B_DIM * S_LEN * D_IN;
        cvt_f32_bf16<<<(n + 255) / 256, 256, 0, stream>>>(x, xb, n);
    }
    build_wcat<<<(G4H * KA + 255) / 256, 256, 0, stream>>>(W_hh_f, W_ih_f, wcatF);
    build_wcat<<<(G4H * KA + 255) / 256, 256, 0, stream>>>(W_hh_b, W_ih_b, wcatB);
    {
        int n = O_DIM * K_OUT;
        cvt_f32_bf16<<<(n + 255) / 256, 256, 0, stream>>>(W_out, woutB, n);
    }

    // 2) bidirectional recurrent scan (grid.x = direction)
    lstm_scan_kernel<<<2, 1024, 0, stream>>>(xb, wcatF, wcatB, b_f, b_b,
                                             h0_f, c0_f, h0_b, c0_b, hsF, hsB);

    // 3) fused output projection + bias + mask
    out_gemm_kernel<<<(B_DIM * S_LEN) / 128, 256, 0, stream>>>(
        hsF, hsB, woutB, b_out, mask, out);
}
